// Block_80152679678600
// MI455X (gfx1250) — matrix operand order repaired and checked by reading, not compiled, not run on hardware
//
#include <hip/hip_runtime.h>
#include <math.h>

// ---------------------------------------------------------------------------
// Types
// ---------------------------------------------------------------------------
typedef __bf16 bf16;
typedef __attribute__((ext_vector_type(16))) __bf16 v16bf;
typedef __attribute__((ext_vector_type(8)))  float  v8f;
typedef __attribute__((ext_vector_type(4)))  float  v4f;

union Frag {
  v16bf v;
  v4f   f4[2];
};

__device__ __forceinline__ bf16 to_bf16(float f) {
  unsigned u = __float_as_uint(f);
  u += 0x7fffu + ((u >> 16) & 1u);           // round-to-nearest-even
  unsigned short h = (unsigned short)(u >> 16);
  return __builtin_bit_cast(bf16, h);
}

// ---------------------------------------------------------------------------
// CDNA5 async global->LDS copy (ASYNCcnt-tracked). 16B per lane.
// ---------------------------------------------------------------------------
__device__ __forceinline__ void async_copy16(const bf16* gsrc, bf16* ldst) {
  // generic pointer to __shared__: low 32 bits are the LDS byte offset
  unsigned loff = (unsigned)(unsigned long long)(void*)ldst;
  asm volatile("global_load_async_to_lds_b128 %0, %1, off"
               :: "v"(loff), "v"(gsrc) : "memory");
}
// wait until at most the newest tile's 4 async ops remain in flight
__device__ __forceinline__ void async_wait4() {
  asm volatile("s_wait_asynccnt 4" ::: "memory");
}
__device__ __forceinline__ void async_wait0() {
  asm volatile("s_wait_asynccnt 0" ::: "memory");
}

// ---------------------------------------------------------------------------
// Weight fp32 [K][N] -> bf16 transposed [N][K]; block (32,8), grid (N/32,K/32)
// ---------------------------------------------------------------------------
__global__ __launch_bounds__(256) void convT_kernel(
    const float* __restrict__ W, bf16* __restrict__ Wt, int K, int N) {
  __shared__ float sm[32][33];
  const int n0 = blockIdx.x * 32, k0 = blockIdx.y * 32;
  const int tx = threadIdx.x, ty = threadIdx.y;
#pragma unroll
  for (int j = 0; j < 4; j++)
    sm[ty + j * 8][tx] = W[(size_t)(k0 + ty + j * 8) * N + n0 + tx];
  __syncthreads();
#pragma unroll
  for (int j = 0; j < 4; j++)
    Wt[(size_t)(n0 + ty + j * 8) * K + k0 + tx] = to_bf16(sm[tx][ty + j * 8]);
}

// ---------------------------------------------------------------------------
// LayerNorm over D=1024, one row per block (256 threads), bf16 output
// ---------------------------------------------------------------------------
__global__ __launch_bounds__(256) void ln_bf16_kernel(
    const float* __restrict__ X, const float* __restrict__ gamma,
    const float* __restrict__ beta, bf16* __restrict__ out) {
  const size_t row = blockIdx.x;
  const float* x = X + row * 1024;
  const int tid = threadIdx.x;
  v4f v = *(const v4f*)(x + tid * 4);
  float s = v[0] + v[1] + v[2] + v[3];
  float q = v[0] * v[0] + v[1] * v[1] + v[2] * v[2] + v[3] * v[3];
#pragma unroll
  for (int m = 1; m < 32; m <<= 1) {
    s += __shfl_xor(s, m, 32);
    q += __shfl_xor(q, m, 32);
  }
  __shared__ float rs[8], rq[8];
  if ((tid & 31) == 0) { rs[tid >> 5] = s; rq[tid >> 5] = q; }
  __syncthreads();
  float S = 0.f, Q = 0.f;
#pragma unroll
  for (int i = 0; i < 8; i++) { S += rs[i]; Q += rq[i]; }
  const float mu  = S * (1.0f / 1024.0f);
  const float var = Q * (1.0f / 1024.0f) - mu * mu;
  const float inv = rsqrtf(var + 1e-5f);
  bf16* o = out + row * 1024;
#pragma unroll
  for (int j = 0; j < 4; j++) {
    const int idx = tid * 4 + j;
    o[idx] = to_bf16((v[j] - mu) * inv * gamma[idx] + beta[idx]);
  }
}

// ---------------------------------------------------------------------------
// WMMA GEMM:  C[M,N] = A[M,K](bf16) x Wt[N,K]^T(bf16) + bias (+resid)(+gelu)
// BM=128 BN=128 BK=32; 8 waves in 2x4 grid, wave tile 64x32 (4x2 wmma frags);
// triple-buffered async global->LDS pipeline, 2 tiles in flight.
// ---------------------------------------------------------------------------
#define LDT 40                    // padded LDS row: 32 bf16 + 8 pad
__global__ __launch_bounds__(256) void gemm_kernel(
    const bf16* __restrict__ A, const bf16* __restrict__ Wt,
    const float* __restrict__ bias, const float* __restrict__ resid,
    float* __restrict__ outF, bf16* __restrict__ outB,
    int M, int N, int K, int doGelu) {
  __shared__ __align__(16) bf16 smA[3][128 * LDT];
  __shared__ __align__(16) bf16 smB[3][128 * LDT];
  const int tid = threadIdx.x;
  const int lane = tid & 31, wid = tid >> 5;
  const int wm = wid >> 2, wn = wid & 3;       // 2 x 4 wave grid
  const int hl = lane & 15, hi = lane >> 4;
  const size_t m0 = (size_t)blockIdx.y * 128;
  const size_t n0 = (size_t)blockIdx.x * 128;
  const int nk = K >> 5;

  v8f acc[4][2] = {};

  auto issue = [&](int kt, int buf) {          // exactly 4 async ops / thread
    const size_t k0 = (size_t)kt * 32;
#pragma unroll
    for (int i = 0; i < 2; i++) {              // A tile: 512 x 16B chunks
      const int c = tid + i * 256;
      const int row = c >> 2, sub = c & 3;
      async_copy16(A + (m0 + row) * K + k0 + sub * 8,
                   &smA[buf][row * LDT + sub * 8]);
    }
#pragma unroll
    for (int i = 0; i < 2; i++) {              // B tile: 512 x 16B chunks
      const int c = tid + i * 256;
      const int row = c >> 2, sub = c & 3;
      async_copy16(Wt + (n0 + row) * K + k0 + sub * 8,
                   &smB[buf][row * LDT + sub * 8]);
    }
  };

  issue(0, 0);                                  // 2 tiles in flight
  issue(1, 1);

  int cur = 0;
  for (int kt = 0; kt < nk; ++kt) {
    if (kt + 1 < nk) async_wait4();             // tile kt landed (in-order)
    else             async_wait0();
    __syncthreads();                            // all waves' copies visible

    if (kt + 2 < nk) {                          // refill the buffer freed at kt-1
      const int nxt = (cur + 2 >= 3) ? cur - 1 : cur + 2;
      issue(kt + 2, nxt);
    }

    Frag af[4], bfm[2];
#pragma unroll
    for (int im = 0; im < 4; im++) {            // A frag: lane=M row, K by lane-half
      const bf16* pA = &smA[cur][(wm * 64 + im * 16 + hl) * LDT];
      af[im].f4[0] = *(const v4f*)(pA + hi * 8);
      af[im].f4[1] = *(const v4f*)(pA + hi * 8 + 16);
    }
#pragma unroll
    for (int in = 0; in < 2; in++) {            // B frag: lane=N col, K by lane-half
      const bf16* pB = &smB[cur][(wn * 32 + in * 16 + hl) * LDT + hi * 8];
      bfm[in].f4[0] = *(const v4f*)(pB);
      bfm[in].f4[1] = *(const v4f*)(pB + 16);
    }
#pragma unroll
    for (int im = 0; im < 4; im++)
#pragma unroll
      for (int in = 0; in < 2; in++)
        acc[im][in] = __builtin_amdgcn_wmma_f32_16x16x32_bf16(
            false, af[im].v, false, bfm[in].v, (short)0, acc[im][in],
            false, false);

    cur = (cur + 1 == 3) ? 0 : cur + 1;
  }

#pragma unroll
  for (int im = 0; im < 4; im++)
#pragma unroll
    for (int in = 0; in < 2; in++)
#pragma unroll
      for (int r = 0; r < 8; r++) {
        const size_t m = m0 + wm * 64 + im * 16 + r + hi * 8;
        const size_t n = n0 + wn * 32 + in * 16 + hl;
        float val = acc[im][in][r] + bias[n];
        if (resid)  val += resid[m * N + n];
        if (doGelu) val = 0.5f * val * (1.0f + erff(val * 0.70710678118654752f));
        if (outF)  outF[m * N + n] = val;
        if (outB)  outB[m * N + n] = to_bf16(val);
      }
}

// ---------------------------------------------------------------------------
// Per-token head-mixing attention: one wave per token.
// scores = (q 16x64)(k 16x64)^T * 1/8 ; softmax over g ; o = P (16x16) V (16x64)
// ---------------------------------------------------------------------------
__global__ __launch_bounds__(256) void attn_kernel(
    const bf16* __restrict__ qkv, bf16* __restrict__ O) {
  __shared__ __align__(16) bf16 smP[8][16][16];
  const int wid = threadIdx.x >> 5, lane = threadIdx.x & 31;
  const int hl = lane & 15, hi = lane >> 4;
  const size_t t = (size_t)blockIdx.x * 8 + wid;
  const bf16* q = qkv + t * 3072;
  const bf16* k = q + 1024;
  const bf16* v = q + 2048;

  Frag qf[2], kf[2];
#pragma unroll
  for (int kk = 0; kk < 2; kk++) {
    const bf16* qp = q + hl * 64 + kk * 32;            // A: row h, d-contig
    qf[kk].f4[0] = *(const v4f*)(qp + hi * 8);
    qf[kk].f4[1] = *(const v4f*)(qp + hi * 8 + 16);
    const bf16* kp = k + hl * 64 + kk * 32 + hi * 8;   // B = k^T: col g, d-contig
    kf[kk].f4[0] = *(const v4f*)(kp);
    kf[kk].f4[1] = *(const v4f*)(kp + 16);
  }
  v8f c = {};
  c = __builtin_amdgcn_wmma_f32_16x16x32_bf16(false, qf[0].v, false, kf[0].v,
                                              (short)0, c, false, false);
  c = __builtin_amdgcn_wmma_f32_16x16x32_bf16(false, qf[1].v, false, kf[1].v,
                                              (short)0, c, false, false);

  // softmax across g: g lives across the 16 lanes of each lane-half
#pragma unroll
  for (int r = 0; r < 8; r++) {
    float sc = c[r] * 0.125f;
    float mx = sc;
#pragma unroll
    for (int m = 1; m < 16; m <<= 1) mx = fmaxf(mx, __shfl_xor(mx, m, 32));
    float e = __expf(sc - mx);
    float sm = e;
#pragma unroll
    for (int m = 1; m < 16; m <<= 1) sm += __shfl_xor(sm, m, 32);
    smP[wid][r + hi * 8][hl] = to_bf16(e / sm);        // (h, g)
  }
  __syncthreads();

  // P as A-matrix frag (K padded 16->32 with zeros)
  Frag pa;
  pa.f4[0] = *(const v4f*)(&smP[wid][hl][hi * 8]);
  pa.f4[1] = (v4f){0.f, 0.f, 0.f, 0.f};

  v8f oacc[4] = {};
#pragma unroll
  for (int cch = 0; cch < 4; cch++) {
    Frag vf;                                            // B = V: col d, g along K
    vf.f4[1] = (v4f){0.f, 0.f, 0.f, 0.f};
#pragma unroll
    for (int e = 0; e < 8; e++) vf.v[e] = v[(size_t)(hi * 8 + e) * 64 + cch * 16 + hl];
    oacc[cch] = __builtin_amdgcn_wmma_f32_16x16x32_bf16(
        false, pa.v, false, vf.v, (short)0, oacc[cch], false, false);
  }

  bf16* orow = O + t * 1024;
#pragma unroll
  for (int cch = 0; cch < 4; cch++)
#pragma unroll
    for (int r = 0; r < 8; r++) {
      const int h = r + hi * 8, d = cch * 16 + hl;
      orow[h * 64 + d] = to_bf16(oacc[cch][r]);
    }
}

// ---------------------------------------------------------------------------
// Host-side orchestration
// ---------------------------------------------------------------------------
extern "C" void kernel_launch(void* const* d_in, const int* in_sizes, int n_in,
                              void* d_out, int out_size, void* d_ws, size_t ws_size,
                              hipStream_t stream) {
  (void)in_sizes; (void)n_in; (void)out_size; (void)ws_size;
  constexpr int M = 8 * 2048;   // 16384 token rows
  constexpr int D = 1024, H3 = 3072, HID = 4096;

  const float* x      = (const float*)d_in[0];
  const float* w_qkv  = (const float*)d_in[1];
  const float* b_qkv  = (const float*)d_in[2];
  const float* w_proj = (const float*)d_in[3];
  const float* b_proj = (const float*)d_in[4];
  const float* w_fc1  = (const float*)d_in[5];
  const float* b_fc1  = (const float*)d_in[6];
  const float* w_fc2  = (const float*)d_in[7];
  const float* b_fc2  = (const float*)d_in[8];
  const float* g1  = (const float*)d_in[9];
  const float* be1 = (const float*)d_in[10];
  const float* g2  = (const float*)d_in[11];
  const float* be2 = (const float*)d_in[12];
  float* out = (float*)d_out;

  char* p = (char*)d_ws;
  auto alloc = [&](size_t elems) -> bf16* {
    bf16* r = (bf16*)p;
    p += ((elems * sizeof(bf16)) + 255) & ~(size_t)255;
    return r;
  };
  bf16* wqkvT  = alloc((size_t)D * H3);
  bf16* wprojT = alloc((size_t)D * D);
  bf16* wfc1T  = alloc((size_t)D * HID);
  bf16* wfc2T  = alloc((size_t)HID * D);
  bf16* h_bf   = alloc((size_t)M * D);
  bf16* o_bf   = alloc((size_t)M * D);
  bf16* big    = alloc((size_t)M * HID);   // qkv (M x 3072) then gelu-act (M x 4096)

  const dim3 tb(32, 8);
  convT_kernel<<<dim3(H3 / 32, D / 32), tb, 0, stream>>>(w_qkv, wqkvT, D, H3);
  convT_kernel<<<dim3(D / 32, D / 32), tb, 0, stream>>>(w_proj, wprojT, D, D);
  convT_kernel<<<dim3(HID / 32, D / 32), tb, 0, stream>>>(w_fc1, wfc1T, D, HID);
  convT_kernel<<<dim3(D / 32, HID / 32), tb, 0, stream>>>(w_fc2, wfc2T, HID, D);

  ln_bf16_kernel<<<M, 256, 0, stream>>>(x, g1, be1, h_bf);
  gemm_kernel<<<dim3(H3 / 128, M / 128), 256, 0, stream>>>(
      h_bf, wqkvT, b_qkv, nullptr, nullptr, big, M, H3, D, 0);
  attn_kernel<<<M / 8, 256, 0, stream>>>(big, o_bf);
  gemm_kernel<<<dim3(D / 128, M / 128), 256, 0, stream>>>(
      o_bf, wprojT, b_proj, x, out, nullptr, M, D, D, 0);
  ln_bf16_kernel<<<M, 256, 0, stream>>>(out, g2, be2, h_bf);
  gemm_kernel<<<dim3(HID / 128, M / 128), 256, 0, stream>>>(
      h_bf, wfc1T, b_fc1, nullptr, nullptr, big, M, HID, D, 1);
  gemm_kernel<<<dim3(D / 128, M / 128), 256, 0, stream>>>(
      big, wfc2T, b_fc2, out, out, nullptr, M, D, HID, 0);
}
